// RegionPredictor_49881750176288
// MI455X (gfx1250) — matrix operand order repaired and checked by reading, not compiled, not run on hardware
//
#include <hip/hip_runtime.h>
#include <hip/hip_bf16.h>
#include <stdint.h>

// ---------------------------------------------------------------------------
// Types for CDNA5 WMMA (wave32): A/B = 16 bf16 per lane, C/D = 8 f32 per lane.
// ---------------------------------------------------------------------------
typedef __attribute__((ext_vector_type(16))) __bf16 v16bf;
typedef __attribute__((ext_vector_type(8)))  float  v8f;

union Frag32 { v16bf v; uint4 q[2]; };   // 32 bytes

#define BN_EPS 1e-5f

// ---------------------------------------------------------------------------
// AntiAliasInterpolation2d: 13x13 gaussian (sigma=1.5), pad 6, subsample /4.
// Output NHWC bf16 with C padded 3 -> 32 (zeros) so conv1 K-chunks are 32-wide.
// ---------------------------------------------------------------------------
__global__ void gauss_down_kernel(const float* __restrict__ x, __bf16* __restrict__ out)
{
    int t = blockIdx.x * blockDim.x + threadIdx.x;
    const int HW = 96 * 96;
    if (t >= 32 * HW) return;
    int b = t / HW; int r = t - b * HW; int oy = r / 96; int ox = r - oy * 96;

    float g[13]; float gs = 0.f;
    #pragma unroll
    for (int i = 0; i < 13; ++i) { float d = (float)i - 6.f; g[i] = __expf(-(d * d) / 4.5f); gs += g[i]; }
    float inv = 1.f / (gs * gs);

    float a0 = 0.f, a1 = 0.f, a2 = 0.f;
    int cy = 4 * oy, cx = 4 * ox;
    for (int i = 0; i < 13; ++i) {
        int sy = cy + i - 6; if (sy < 0 || sy >= 384) continue;
        for (int j = 0; j < 13; ++j) {
            int sx = cx + j - 6; if (sx < 0 || sx >= 384) continue;
            float kw = g[i] * g[j];
            a0 += kw * x[(((size_t)b * 3 + 0) * 384 + sy) * 384 + sx];
            a1 += kw * x[(((size_t)b * 3 + 1) * 384 + sy) * 384 + sx];
            a2 += kw * x[(((size_t)b * 3 + 2) * 384 + sy) * 384 + sx];
        }
    }
    __bf16* o = out + (size_t)t * 32;
    o[0] = (__bf16)(a0 * inv); o[1] = (__bf16)(a1 * inv); o[2] = (__bf16)(a2 * inv);
    #pragma unroll
    for (int c = 3; c < 32; ++c) o[c] = (__bf16)0.f;
}

// ---------------------------------------------------------------------------
// Pack OIHW fp32 weights -> bf16 [kk][cinChunk][coutPad][32] so each B-frag
// lane reads 16 contiguous K values (k = 16*half + s, same order as the A gather).
// ---------------------------------------------------------------------------
__global__ void pack_w_kernel(const float* __restrict__ w, __bf16* __restrict__ wp,
                              int Cout, int Cin, int K, int CC, int CoutPad)
{
    size_t total = (size_t)K * K * CC * CoutPad * 32;
    size_t i = (size_t)blockIdx.x * blockDim.x + threadIdx.x;
    if (i >= total) return;
    int kc = (int)(i & 31);
    size_t t = i >> 5;
    int n  = (int)(t % CoutPad); t /= CoutPad;
    int cc = (int)(t % CC);      t /= CC;
    int kk = (int)t;
    int ky = kk / K, kx = kk - ky * K;
    int cin = cc * 32 + kc;
    float v = 0.f;
    if (n < Cout && cin < Cin)
        v = w[(((size_t)n * Cin + cin) * K + ky) * K + kx];
    wp[i] = (__bf16)v;
}

__global__ void fill0_kernel(float* p, int n)
{
    int i = blockIdx.x * blockDim.x + threadIdx.x;
    if (i < n) p[i] = 0.f;
}

// ---------------------------------------------------------------------------
// Implicit-GEMM conv (stride 1, odd K, same-pad) via v_wmma_f32_16x16x32_bf16.
// One wave = 16(pixels) x NT*16(channels): one A fragment is reused across NT
// B fragments / WMMAs per K-step (4x arithmetic intensity for NT=4).
// Optional nearest 2x upsample folded into the gather; optional BN stats
// (per-channel sum/sumsq) via cross-half shuffle + atomics; bias fused.
// ---------------------------------------------------------------------------
template <int NT>
__global__ void conv_wmma_kernel(const __bf16* __restrict__ in,
                                 const __bf16* __restrict__ wp,
                                 const float* __restrict__ bias,
                                 void* __restrict__ out,
                                 float* __restrict__ stats,
                                 int Bn, int H, int W,
                                 int CinPad, int Cout, int CoutPad,
                                 int K, int upsample, int fp32out)
{
    const int lane = threadIdx.x & 31;
    const int wave = blockIdx.x * (blockDim.x >> 5) + (threadIdx.x >> 5);
    const int P = Bn * H * W;
    const int mtiles = P >> 4;
    const int ngroups = CoutPad / (16 * NT);
    if (wave >= mtiles * ngroups) return;          // wave-uniform: EXEC stays all-ones
    const int ng = wave % ngroups;
    const int mt = wave / ngroups;
    const int CC = CinPad >> 5;
    const int pad = K >> 1;
    const int Hs = upsample ? (H >> 1) : H;
    const int Ws = upsample ? (W >> 1) : W;

    // A-fragment pixel for this lane (lane and lane+16 share the same row M).
    const int p  = mt * 16 + (lane & 15);
    const int b  = p / (H * W);
    const int rp = p - b * (H * W);
    const int y  = rp / W;
    const int x  = rp - y * W;
    const int hi  = lane >> 4;
    const int ai0 = hi ? 2 : 0;   // K 0-7   / 16-23  (16B chunk index in 64B cin-chunk)
    const int ai1 = hi ? 3 : 1;   // K 8-15  / 24-31

    // Lane's weight offset (uint4 units) for tile t=0 at (kk,cc)=(0,0).
    const size_t wlane = ((size_t)ng * NT * 16 + (lane & 15)) * 4 + (hi ? 2 : 0);

    v8f zero = {};
    v8f acc[NT];
    #pragma unroll
    for (int t = 0; t < NT; ++t) acc[t] = zero;

    for (int ky = 0; ky < K; ++ky) {
        const int iy = y + ky - pad;
        const bool vy = (iy >= 0) & (iy < H);
        const int sy = upsample ? (iy >> 1) : iy;
        const __bf16* rowp = in + (size_t)(b * Hs + sy) * Ws * CinPad;
        for (int kx = 0; kx < K; ++kx) {
            const int ix = x + kx - pad;
            const bool vv = vy & (ix >= 0) & (ix < W);
            const int sx = upsample ? (ix >> 1) : ix;
            const int kk = ky * K + kx;
            const __bf16* srcp = rowp + (size_t)sx * CinPad;
            const uint4* wbase = (const uint4*)wp + (size_t)kk * CC * CoutPad * 4 + wlane;
            if (vv) __builtin_prefetch(srcp + CinPad, 0, 1);   // global_prefetch_b8 (next pixel)
            for (int cc = 0; cc < CC; ++cc) {
                Frag32 a;
                if (vv) {
                    const uint4* s4 = (const uint4*)(srcp + cc * 32);
                    a.q[0] = s4[ai0];
                    a.q[1] = s4[ai1];
                } else {
                    a.q[0] = make_uint4(0u, 0u, 0u, 0u);
                    a.q[1] = make_uint4(0u, 0u, 0u, 0u);
                }
                const uint4* w4 = wbase + (size_t)cc * CoutPad * 4;
                #pragma unroll
                for (int t = 0; t < NT; ++t) {
                    Frag32 bf;
                    bf.q[0] = w4[t * 64 + 0];   // tile t = +16 channels = +64 uint4
                    bf.q[1] = w4[t * 64 + 1];
                    acc[t] = __builtin_amdgcn_wmma_f32_16x16x32_bf16(
                                 false, a.v, false, bf.v, (short)0, acc[t], false, false);
                }
            }
        }
    }

    // C/D layout: lane = channel (lane&15) of tile; VGPR r = pixel r (+8 for hi half)
    #pragma unroll
    for (int t = 0; t < NT; ++t) {
        const int ch = ng * NT * 16 + t * 16 + (lane & 15);
        const float bv = (ch < Cout) ? bias[ch] : 0.f;
        float vals[8];
        float s1 = 0.f, s2 = 0.f;
        #pragma unroll
        for (int r = 0; r < 8; ++r) { float v = acc[t][r] + bv; vals[r] = v; s1 += v; s2 += v * v; }

        if (stats) {
            s1 += __shfl_xor(s1, 16);
            s2 += __shfl_xor(s2, 16);
            if (hi == 0 && ch < Cout) {
                atomicAdd(&stats[ch], s1);
                atomicAdd(&stats[1024 + ch], s2);
            }
        }
        #pragma unroll
        for (int r = 0; r < 8; ++r) {
            const int pm = mt * 16 + r + (hi ? 8 : 0);
            const size_t idx = (size_t)pm * CoutPad + ch;
            if (fp32out) ((float*)out)[idx] = vals[r];
            else         ((__bf16*)out)[idx] = (__bf16)vals[r];
        }
    }
}

// ---------------------------------------------------------------------------
// BN (training-mode): scale = gamma*rsqrt(var+eps), shift = beta - mean*scale
// ---------------------------------------------------------------------------
__global__ void bn_finalize_kernel(const float* __restrict__ stats,
                                   const float* __restrict__ gamma,
                                   const float* __restrict__ beta,
                                   float* __restrict__ ss, int Cout, float invN)
{
    int c = blockIdx.x * blockDim.x + threadIdx.x;
    if (c >= Cout) return;
    float m   = stats[c] * invN;
    float var = stats[1024 + c] * invN - m * m;
    float sc  = gamma[c] * rsqrtf(var + BN_EPS);
    ss[c] = sc;
    ss[1024 + c] = beta[c] - m * sc;
}

// Down block tail: BN + ReLU + 2x2 avgpool
__global__ void bn_relu_pool_kernel(const __bf16* __restrict__ pre, const float* __restrict__ ss,
                                    __bf16* __restrict__ out, int Bn, int H, int W, int C)
{
    const int Ho = H >> 1, Wo = W >> 1;
    size_t total = (size_t)Bn * Ho * Wo * C;
    size_t i = (size_t)blockIdx.x * blockDim.x + threadIdx.x;
    if (i >= total) return;
    int c = (int)(i % C); size_t pt = i / C;
    int xo = (int)(pt % Wo); size_t t2 = pt / Wo;
    int yo = (int)(t2 % Ho); int b = (int)(t2 / Ho);
    float sc = ss[c], sh = ss[1024 + c];
    float s = 0.f;
    #pragma unroll
    for (int dy = 0; dy < 2; ++dy)
        #pragma unroll
        for (int dx = 0; dx < 2; ++dx) {
            float v = (float)pre[((size_t)(b * H + 2 * yo + dy) * W + 2 * xo + dx) * C + c] * sc + sh;
            s += fmaxf(v, 0.f);
        }
    out[i] = (__bf16)(0.25f * s);
}

// Up block tail: BN + ReLU written into concat buffer (channel offset coff)
__global__ void bn_relu_cat_kernel(const __bf16* __restrict__ pre, const float* __restrict__ ss,
                                   __bf16* __restrict__ cat, size_t P, int C, int Ccat, int coff)
{
    size_t total = P * C;
    size_t i = (size_t)blockIdx.x * blockDim.x + threadIdx.x;
    if (i >= total) return;
    int c = (int)(i % C); size_t p = i / C;
    float v = (float)pre[i] * ss[c] + ss[1024 + c];
    cat[p * (size_t)Ccat + coff + c] = (__bf16)fmaxf(v, 0.f);
}

// Skip copy into concat buffer; channels >= Cs are zero-filled padding.
__global__ void copy_skip_kernel(const __bf16* __restrict__ skip, __bf16* __restrict__ cat,
                                 size_t P, int Cs, int Cstride, int Ccat, int coff, int nch)
{
    size_t total = P * nch;
    size_t i = (size_t)blockIdx.x * blockDim.x + threadIdx.x;
    if (i >= total) return;
    int c = (int)(i % nch); size_t p = i / nch;
    __bf16 v = (c < Cs) ? skip[p * (size_t)Cstride + c] : (__bf16)0.f;
    cat[p * (size_t)Ccat + coff + c] = v;
}

// ---------------------------------------------------------------------------
// Per-(b,k): spatial softmax (T=0.1), write region, compute shift + covar
// (covar = E[g g^T] - shift shift^T since weights sum to 1).
// ---------------------------------------------------------------------------
__global__ void region_stats_kernel(const float* __restrict__ pred,
                                    float* __restrict__ region,
                                    float* __restrict__ shift,
                                    float* __restrict__ covar,
                                    int Kr, int H, int W, int Cp)
{
    __shared__ float red[256];
    const int bk = blockIdx.x;
    const int b = bk / Kr, k = bk - b * Kr;
    const int tid = threadIdx.x;
    const int P = H * W;
    const float invT = 10.f;   // 1 / 0.1
    const float* pb = pred + (size_t)b * P * Cp + k;

    float mx = -3.402823e38f;
    for (int i = tid; i < P; i += 256) mx = fmaxf(mx, pb[(size_t)i * Cp] * invT);
    red[tid] = mx; __syncthreads();
    for (int s = 128; s > 0; s >>= 1) { if (tid < s) red[tid] = fmaxf(red[tid], red[tid + s]); __syncthreads(); }
    mx = red[0]; __syncthreads();

    float sum = 0.f;
    for (int i = tid; i < P; i += 256) sum += __expf(pb[(size_t)i * Cp] * invT - mx);
    red[tid] = sum; __syncthreads();
    for (int s = 128; s > 0; s >>= 1) { if (tid < s) red[tid] += red[tid + s]; __syncthreads(); }
    const float inv = 1.f / red[0]; __syncthreads();

    float sx = 0.f, sy = 0.f, sxx = 0.f, sxy = 0.f, syy = 0.f;
    const float wsx = 2.f / (float)(W - 1), wsy = 2.f / (float)(H - 1);
    for (int i = tid; i < P; i += 256) {
        float w = __expf(pb[(size_t)i * Cp] * invT - mx) * inv;
        int iy = i / W, ix = i - iy * W;
        float gx = wsx * ix - 1.f, gy = wsy * iy - 1.f;
        region[(size_t)bk * P + i] = w;
        sx += w * gx; sy += w * gy;
        sxx += w * gx * gx; sxy += w * gx * gy; syy += w * gy * gy;
    }
    float vals5[5] = { sx, sy, sxx, sxy, syy };
    float out5[5];
    for (int j = 0; j < 5; ++j) {
        red[tid] = vals5[j]; __syncthreads();
        for (int s = 128; s > 0; s >>= 1) { if (tid < s) red[tid] += red[tid + s]; __syncthreads(); }
        out5[j] = red[0]; __syncthreads();
    }
    if (tid == 0) {
        shift[(size_t)bk * 2 + 0] = out5[0];
        shift[(size_t)bk * 2 + 1] = out5[1];
        float cxx = out5[2] - out5[0] * out5[0];
        float cxy = out5[3] - out5[0] * out5[1];
        float cyy = out5[4] - out5[1] * out5[1];
        covar[(size_t)bk * 4 + 0] = cxx; covar[(size_t)bk * 4 + 1] = cxy;
        covar[(size_t)bk * 4 + 2] = cxy; covar[(size_t)bk * 4 + 3] = cyy;
    }
}

// ---------------------------------------------------------------------------
// Closed-form SVD of symmetric PSD 2x2 (== eigendecomposition, sigma desc.):
// u columns = eigenvectors, d = diag(sqrt(s)), affine = u @ d.
// ---------------------------------------------------------------------------
__global__ void svd_affine_kernel(const float* __restrict__ covar,
                                  float* __restrict__ affine,
                                  float* __restrict__ u_out,
                                  float* __restrict__ d_outp, int N)
{
    int i = blockIdx.x * blockDim.x + threadIdx.x;
    if (i >= N) return;
    float a  = covar[(size_t)i * 4 + 0];
    float bb = covar[(size_t)i * 4 + 1];
    float c  = covar[(size_t)i * 4 + 3];
    float tr2 = 0.5f * (a + c);
    float db  = 0.5f * (a - c);
    float disc = sqrtf(db * db + bb * bb);
    float l0 = tr2 + disc;
    float l1 = fmaxf(tr2 - disc, 0.f);
    float vx, vy;
    if (fabsf(bb) > 1e-30f)      { vx = bb;  vy = l0 - a; }
    else if (a >= c)             { vx = 1.f; vy = 0.f; }
    else                         { vx = 0.f; vy = 1.f; }
    float nrm = sqrtf(vx * vx + vy * vy);
    if (nrm > 0.f) { vx /= nrm; vy /= nrm; }
    float s0 = sqrtf(fmaxf(l0, 0.f));
    float s1 = sqrtf(l1);
    u_out[(size_t)i * 4 + 0] = vx;  u_out[(size_t)i * 4 + 1] = -vy;
    u_out[(size_t)i * 4 + 2] = vy;  u_out[(size_t)i * 4 + 3] = vx;
    d_outp[(size_t)i * 4 + 0] = s0; d_outp[(size_t)i * 4 + 1] = 0.f;
    d_outp[(size_t)i * 4 + 2] = 0.f; d_outp[(size_t)i * 4 + 3] = s1;
    affine[(size_t)i * 4 + 0] = vx * s0; affine[(size_t)i * 4 + 1] = -vy * s1;
    affine[(size_t)i * 4 + 2] = vy * s0; affine[(size_t)i * 4 + 3] = vx * s1;
}

// ---------------------------------------------------------------------------
// Host driver
// ---------------------------------------------------------------------------
static inline size_t align256(size_t x) { return (x + 255) & ~(size_t)255; }

extern "C" void kernel_launch(void* const* d_in, const int* in_sizes, int n_in,
                              void* d_out, int out_size, void* d_ws, size_t ws_size,
                              hipStream_t stream)
{
    (void)in_sizes; (void)n_in; (void)out_size; (void)ws_size;
    const int B = 32;

    const float* x = (const float*)d_in[0];
    const float *dw[5], *dbi[5], *dg[5], *dbe[5];
    const float *uw[5], *ubi[5], *ug[5], *ube[5];
    for (int i = 0; i < 5; ++i) {
        dw[i]  = (const float*)d_in[1 + 4 * i];  dbi[i] = (const float*)d_in[2 + 4 * i];
        dg[i]  = (const float*)d_in[3 + 4 * i];  dbe[i] = (const float*)d_in[4 + 4 * i];
    }
    for (int i = 0; i < 5; ++i) {
        uw[i]  = (const float*)d_in[21 + 4 * i]; ubi[i] = (const float*)d_in[22 + 4 * i];
        ug[i]  = (const float*)d_in[23 + 4 * i]; ube[i] = (const float*)d_in[24 + 4 * i];
    }
    const float* rw = (const float*)d_in[41];
    const float* rb = (const float*)d_in[42];

    // ---- workspace carve-out (all NHWC bf16, channels padded to mult. 32) ----
    char* ws = (char*)d_ws;
    size_t off = 0;
    auto take = [&](size_t bytes) -> char* { char* p = ws + off; off = align256(off + bytes); return p; };

    __bf16* xaa  = (__bf16*)take((size_t)B * 96 * 96 * 32 * 2);
    void*   pre  = (void*)  take((size_t)B * 96 * 96 * 64 * 2);   // shared pre-act buffer (max 37.7MB)
    __bf16* d1   = (__bf16*)take((size_t)B * 48 * 48 * 64 * 2);
    __bf16* d2   = (__bf16*)take((size_t)B * 24 * 24 * 128 * 2);
    __bf16* d3   = (__bf16*)take((size_t)B * 12 * 12 * 256 * 2);
    __bf16* d4   = (__bf16*)take((size_t)B * 6 * 6 * 512 * 2);
    __bf16* d5   = (__bf16*)take((size_t)B * 3 * 3 * 1024 * 2);
    __bf16* cat1 = (__bf16*)take((size_t)B * 6 * 6 * 1024 * 2);
    __bf16* cat2 = (__bf16*)take((size_t)B * 12 * 12 * 512 * 2);
    __bf16* cat3 = (__bf16*)take((size_t)B * 24 * 24 * 256 * 2);
    __bf16* cat4 = (__bf16*)take((size_t)B * 48 * 48 * 128 * 2);
    __bf16* cat5 = (__bf16*)take((size_t)B * 96 * 96 * 64 * 2);
    float*  stats = (float*)take(2048 * 4);
    float*  ssbuf = (float*)take(2048 * 4);
    __bf16* wpack = (__bf16*)take((size_t)9 * 32 * 1024 * 32 * 2); // shared packed-weight buffer

    auto run_conv = [&](const __bf16* inp, const float* wsrc, const float* bias,
                        int Cin, int CinPad, int Cout, int CoutPad, int Ksz,
                        int H, int W, int ups, void* outp, int fp32o, bool doStats) {
        int CC = CinPad / 32;
        size_t wtot = (size_t)Ksz * Ksz * CC * CoutPad * 32;
        pack_w_kernel<<<(unsigned)((wtot + 255) / 256), 256, 0, stream>>>(
            wsrc, wpack, Cout, Cin, Ksz, CC, CoutPad);
        if (doStats) fill0_kernel<<<8, 256, 0, stream>>>(stats, 2048);
        float* st = doStats ? stats : (float*)nullptr;
        int nt = (CoutPad >= 64) ? 4 : ((CoutPad >= 32) ? 2 : 1);
        int waves = ((B * H * W) / 16) * (CoutPad / (16 * nt));
        int blocks = (waves + 7) / 8;      // 256 threads = 8 waves per block
        switch (nt) {
        case 4:
            conv_wmma_kernel<4><<<blocks, 256, 0, stream>>>(
                inp, wpack, bias, outp, st, B, H, W, CinPad, Cout, CoutPad, Ksz, ups, fp32o);
            break;
        case 2:
            conv_wmma_kernel<2><<<blocks, 256, 0, stream>>>(
                inp, wpack, bias, outp, st, B, H, W, CinPad, Cout, CoutPad, Ksz, ups, fp32o);
            break;
        default:
            conv_wmma_kernel<1><<<blocks, 256, 0, stream>>>(
                inp, wpack, bias, outp, st, B, H, W, CinPad, Cout, CoutPad, Ksz, ups, fp32o);
            break;
        }
    };

    // ---- anti-alias 0.25x downsample ----
    gauss_down_kernel<<<(B * 96 * 96 + 255) / 256, 256, 0, stream>>>(x, xaa);

    // ---- encoder ----
    const __bf16* din[5]  = { xaa, d1, d2, d3, d4 };
    __bf16*       dout[5] = { d1, d2, d3, d4, d5 };
    const int dCin[5]  = { 3, 64, 128, 256, 512 };
    const int dCinP[5] = { 32, 64, 128, 256, 512 };
    const int dCout[5] = { 64, 128, 256, 512, 1024 };
    const int dH[5]    = { 96, 48, 24, 12, 6 };
    for (int i = 0; i < 5; ++i) {
        int H = dH[i];
        run_conv(din[i], dw[i], dbi[i], dCin[i], dCinP[i], dCout[i], dCout[i], 3, H, H, 0, pre, 0, true);
        float invN = 1.f / ((float)B * H * H);
        bn_finalize_kernel<<<(dCout[i] + 255) / 256, 256, 0, stream>>>(stats, dg[i], dbe[i], ssbuf, dCout[i], invN);
        size_t tot = (size_t)B * (H / 2) * (H / 2) * dCout[i];
        bn_relu_pool_kernel<<<(unsigned)((tot + 255) / 256), 256, 0, stream>>>(
            (const __bf16*)pre, ssbuf, dout[i], B, H, H, dCout[i]);
    }

    // ---- decoder (upsample folded into conv gather; concat materialized) ----
    const __bf16* uin[5]   = { d5, cat1, cat2, cat3, cat4 };
    __bf16*       ucat[5]  = { cat1, cat2, cat3, cat4, cat5 };
    const __bf16* uskip[5] = { d4, d3, d2, d1, xaa };
    const int uCin[5]   = { 1024, 1024, 512, 256, 128 };
    const int uCout[5]  = { 512, 256, 128, 64, 32 };
    const int uH[5]     = { 6, 12, 24, 48, 96 };
    const int uCcat[5]  = { 1024, 512, 256, 128, 64 };
    const int uSkipC[5] = { 512, 256, 128, 64, 3 };
    const int uSkipS[5] = { 512, 256, 128, 64, 32 };
    for (int i = 0; i < 5; ++i) {
        int H = uH[i];
        run_conv(uin[i], uw[i], ubi[i], uCin[i], uCin[i], uCout[i], uCout[i], 3, H, H, 1, pre, 0, true);
        float invN = 1.f / ((float)B * H * H);
        bn_finalize_kernel<<<(uCout[i] + 255) / 256, 256, 0, stream>>>(stats, ug[i], ube[i], ssbuf, uCout[i], invN);
        size_t P = (size_t)B * H * H;
        size_t tot = P * uCout[i];
        bn_relu_cat_kernel<<<(unsigned)((tot + 255) / 256), 256, 0, stream>>>(
            (const __bf16*)pre, ssbuf, ucat[i], P, uCout[i], uCcat[i], 0);
        int nch = uCcat[i] - uCout[i];
        size_t tot2 = P * nch;
        copy_skip_kernel<<<(unsigned)((tot2 + 255) / 256), 256, 0, stream>>>(
            uskip[i], ucat[i], P, uSkipC[i], uSkipS[i], uCcat[i], uCout[i], nch);
    }

    // ---- region head: 7x7 conv, Cin 35 (pad 64) -> Cout 10 (pad 16), fp32 out ----
    run_conv(cat5, rw, rb, 35, 64, 10, 16, 7, 96, 96, 0, pre, 1, false);

    // ---- softmax + moments + SVD -> outputs (shift,covar,region,affine,u,d) ----
    float* out = (float*)d_out;
    float* o_shift  = out;                                   // 32*10*2
    float* o_covar  = out + 640;                             // 32*10*4
    float* o_region = out + 1920;                            // 32*10*96*96
    float* o_affine = o_region + (size_t)B * 10 * 96 * 96;   // 32*10*4
    float* o_u      = o_affine + 1280;
    float* o_d      = o_u + 1280;

    region_stats_kernel<<<B * 10, 256, 0, stream>>>((const float*)pre, o_region, o_shift, o_covar, 10, 96, 96, 16);
    svd_affine_kernel<<<2, 256, 0, stream>>>(o_covar, o_affine, o_u, o_d, B * 10);
}
